// BertSelfAttention4_59236188946618
// MI455X (gfx1250) — matrix operand order repaired and checked by reading, not compiled, not run on hardware
//
#include <hip/hip_runtime.h>
#include <hip/hip_bf16.h>
#include <math.h>
#include <stdint.h>

typedef __bf16 bf16;
typedef __attribute__((ext_vector_type(16))) __bf16 v16bf;
typedef __attribute__((ext_vector_type(8)))  __bf16 v8bf;
typedef __attribute__((ext_vector_type(4)))  __bf16 v4bf;
typedef __attribute__((ext_vector_type(8)))  float  v8f;

#define D_MODEL 1024
#define SEQ     2048
#define BATCH   2
#define NHEADS  16
#define HDIM    64

__device__ __forceinline__ v8f wmma_bf16(v16bf a, v16bf b, v8f c) {
  return __builtin_amdgcn_wmma_f32_16x16x32_bf16(
      /*neg_a=*/false, a, /*neg_b=*/false, b,
      /*c_mod=*/(short)0, c, /*reuse_a=*/false, /*reuse_b=*/false);
}

__device__ __forceinline__ v16bf cat8(v8bf lo, v8bf hi) {
  return __builtin_shufflevector(lo, hi, 0, 1, 2, 3, 4, 5, 6, 7,
                                         8, 9, 10, 11, 12, 13, 14, 15);
}

// A-matrix 16x32 bf16 fragment (row-major tile, rows 16B aligned): per lane
// two contiguous 8-elem runs (K = kh..kh+7, kh+16..kh+23) -> 2x ds_load_b128.
__device__ __forceinline__ v16bf load_a_frag(const bf16* p, int ld, int lane) {
  const int m  = lane & 15;
  const int kh = (lane >> 4) << 3;
  const bf16* q = p + m * ld + kh;
  return cat8(*(const v8bf*)(q), *(const v8bf*)(q + 16));
}

// B-matrix 32x16 bf16 fragment from K-CONTIGUOUS storage t[n][k]:
// lane n needs K = kh..kh+7, kh+16..kh+23 -> 2x ds_load_b128.
__device__ __forceinline__ v16bf load_b_fragT(const bf16* p, int ld, int lane) {
  const int n  = lane & 15;
  const int kh = (lane >> 4) << 3;
  const bf16* q = p + n * ld + kh;
  return cat8(*(const v8bf*)(q), *(const v8bf*)(q + 16));
}

// Fast log-sigmoid: hardware v_exp/v_log instead of libm log1pf expansion.
__device__ __forceinline__ float logsig(float x) {
  return fminf(x, 0.0f) - __logf(1.0f + __expf(-fabsf(x)));
}

// Async global -> LDS 16-byte copy (gfx1250, ASYNCcnt-tracked).
// LDS offset = low 32 bits of generic shared pointer (LDS aperture).
__device__ __forceinline__ void async_copy_b128(void* lds_ptr, const void* gptr) {
  unsigned lds_off = (unsigned)(uintptr_t)lds_ptr;
  unsigned long long ga = (unsigned long long)(uintptr_t)gptr;
  asm volatile("global_load_async_to_lds_b128 %0, %1, off"
               :: "v"(lds_off), "v"(ga) : "memory");
}
__device__ __forceinline__ void async_wait0() {
  asm volatile("s_wait_asynccnt 0x0" ::: "memory");
}

// ---------------------------------------------------------------------------
// Stage 1: fused Q/K projection GEMM + logsigmoid epilogue.
//   q  = hidden @ Wq + bq ;  k2 = logsig(logsig(q) + q + hidden @ Wk + bk)
// Writes bf16 q and k2, layout [B*S, D]. Block 128 thr; tile 64x64; BK=32.
// ---------------------------------------------------------------------------
__global__ __launch_bounds__(128) void qk_proj_kernel(
    const float* __restrict__ hidden,
    const float* __restrict__ Wq, const float* __restrict__ bq,
    const float* __restrict__ Wk, const float* __restrict__ bk,
    bf16* __restrict__ qbf, bf16* __restrict__ kbf)
{
  __shared__ __align__(16) bf16 aT [64][40];   // [m][k]
  __shared__ __align__(16) bf16 bqT[64][40];   // [n][k] K-contiguous
  __shared__ __align__(16) bf16 bkT[64][40];   // [n][k]

  const int tid     = threadIdx.x;
  const int lane    = tid & 31;
  const int wave    = tid >> 5;
  const int colBase = blockIdx.x * 64;
  const int rowBase = blockIdx.y * 64;

  v8f accQ[4], accK[4];
#pragma unroll
  for (int s = 0; s < 4; ++s) { accQ[s] = {}; accK[s] = {}; }

#pragma unroll 1
  for (int k0 = 0; k0 < D_MODEL; k0 += 32) {
    // A tile 64x32: float4 global loads -> packed b64 LDS stores.
#pragma unroll
    for (int i = 0; i < 4; ++i) {
      int idx = tid + i * 128;
      int r = idx >> 3, c4 = (idx & 7) << 2;
      float4 f = *(const float4*)(hidden + (size_t)(rowBase + r) * D_MODEL + k0 + c4);
      v4bf t; t[0] = (bf16)f.x; t[1] = (bf16)f.y; t[2] = (bf16)f.z; t[3] = (bf16)f.w;
      *(v4bf*)&aT[r][c4] = t;
    }
    // Weight tiles K-major: 4 lane-coalesced k-strided loads -> one b64 store.
#pragma unroll
    for (int i = 0; i < 4; ++i) {
      int idx = tid + i * 128;
      int n = idx & 63, kq = (idx >> 6) << 2;
      v4bf tq, tk;
#pragma unroll
      for (int j = 0; j < 4; ++j) {
        size_t g = (size_t)(k0 + kq + j) * D_MODEL + colBase + n;
        tq[j] = (bf16)Wq[g];
        tk[j] = (bf16)Wk[g];
      }
      *(v4bf*)&bqT[n][kq] = tq;
      *(v4bf*)&bkT[n][kq] = tk;
    }
    __syncthreads();

    // Batch all fragment loads, then a burst of 8 WMMAs (single dscnt wait).
    v16bf af = load_a_frag(&aT[wave * 16][0], 40, lane);
    v16bf bqf[4], bkf[4];
#pragma unroll
    for (int s = 0; s < 4; ++s) {
      bqf[s] = load_b_fragT(&bqT[s * 16][0], 40, lane);
      bkf[s] = load_b_fragT(&bkT[s * 16][0], 40, lane);
    }
#pragma unroll
    for (int s = 0; s < 4; ++s) accQ[s] = wmma_bf16(af, bqf[s], accQ[s]);
#pragma unroll
    for (int s = 0; s < 4; ++s) accK[s] = wmma_bf16(af, bkf[s], accK[s]);
    __syncthreads();
  }

  const int half = lane >> 4;
  const int n0   = lane & 15;
#pragma unroll
  for (int s = 0; s < 4; ++s) {
    const int col = colBase + s * 16 + n0;
    const float biasQ = bq[col];
    const float biasK = bk[col];
#pragma unroll
    for (int r = 0; r < 8; ++r) {
      const int row = rowBase + wave * 16 + r + 8 * half;
      const float qv = accQ[s][r] + biasQ;
      const float kv = accK[s][r] + biasK;
      const float k2 = logsig(logsig(qv) + qv + kv);
      qbf[(size_t)row * D_MODEL + col] = (bf16)qv;
      kbf[(size_t)row * D_MODEL + col] = (bf16)k2;
    }
  }
}

// ---------------------------------------------------------------------------
// Stage 2: flash-style attention per (b, h, 64-query tile).
//   scores = -(q . k2)/8, query-row mask, online softmax, ctx = P @ q
// ---------------------------------------------------------------------------
__global__ __launch_bounds__(128) void attn_kernel(
    const bf16* __restrict__ qbf, const bf16* __restrict__ kbf,
    const int* __restrict__ amask, float* __restrict__ out)
{
  __shared__ __align__(16) bf16 kTs[64][72];    // [key][d]  (scores B)
  __shared__ __align__(16) bf16 vTt[64][72];    // [d][key]  (PV B, K-contig)
  __shared__ __align__(16) bf16 pT[4][16][72];  // per-wave P, [m][key]

  const int tid  = threadIdx.x;
  const int lane = tid & 31;
  const int wave = tid >> 5;
  const int qt   = blockIdx.x;
  const int h    = blockIdx.y;
  const int b    = blockIdx.z;
  const int qBase = qt * 64 + wave * 16;
  const size_t rowOff = (size_t)b * SEQ;
  const int cOff = h * HDIM;

  // Q A-fragments straight from global (4 x 16-byte loads per lane).
  v16bf qA[2];
  {
    const int m  = lane & 15;
    const int kh = (lane >> 4) << 3;
    const bf16* q = qbf + (rowOff + qBase + m) * D_MODEL + cOff;
    qA[0] = cat8(*(const v8bf*)(q + kh),      *(const v8bf*)(q + kh + 16));
    qA[1] = cat8(*(const v8bf*)(q + 32 + kh), *(const v8bf*)(q + 32 + kh + 16));
  }

  const int half = lane >> 4;
  bool  mOK[8];
  float mSt[8], lSt[8];
  v8f   accO[4];
#pragma unroll
  for (int s = 0; s < 4; ++s) accO[s] = {};
#pragma unroll
  for (int r = 0; r < 8; ++r) {
    mSt[r] = -1e30f;
    lSt[r] = 0.0f;
    mOK[r] = amask[b * SEQ + qBase + r + 8 * half] != 0;
  }

#pragma unroll 1
  for (int kt = 0; kt < SEQ; kt += 64) {
    // K2 tile [key][d]: async global->LDS b128 copies (ASYNCcnt).
#pragma unroll
    for (int i = 0; i < 4; ++i) {
      int idx = tid + i * 128;
      int r = idx >> 3, c8 = (idx & 7) << 3;
      async_copy_b128(&kTs[r][c8],
                      kbf + (rowOff + kt + r) * D_MODEL + cOff + c8);
    }
    // V(=Q) tile transposed [d][key]: lane-coalesced key-strided loads.
#pragma unroll
    for (int i = 0; i < 8; ++i) {
      int idx = tid + i * 128;
      int d = idx & 63, kq = (idx >> 6) << 2;
      v4bf t;
#pragma unroll
      for (int j = 0; j < 4; ++j)
        t[j] = qbf[(rowOff + kt + kq + j) * D_MODEL + cOff + d];
      *(v4bf*)&vTt[d][kq] = t;
    }
    if (kt + 64 < SEQ)
      __builtin_prefetch(qbf + (rowOff + kt + 64 + (tid >> 1)) * D_MODEL + cOff, 0, 1);
    async_wait0();
    __syncthreads();

    // scores = -(Q . K2^T)/8 : batch 8 B-frag loads, then 8 WMMAs.
    v16bf kb[4][2];
#pragma unroll
    for (int s = 0; s < 4; ++s) {
      kb[s][0] = load_b_fragT(&kTs[s * 16][0],  72, lane);
      kb[s][1] = load_b_fragT(&kTs[s * 16][32], 72, lane);
    }
    v8f accS[4];
#pragma unroll
    for (int s = 0; s < 4; ++s) { accS[s] = {}; accS[s] = wmma_bf16(qA[0], kb[s][0], accS[s]); }
#pragma unroll
    for (int s = 0; s < 4; ++s) accS[s] = wmma_bf16(qA[1], kb[s][1], accS[s]);

    float sc[4][8];
    float mNew[8];
#pragma unroll
    for (int r = 0; r < 8; ++r) mNew[r] = mSt[r];
#pragma unroll
    for (int s = 0; s < 4; ++s)
#pragma unroll
      for (int r = 0; r < 8; ++r) {
        float v = -0.125f * accS[s][r];
        sc[s][r] = mOK[r] ? v : -10000.0f;
        mNew[r]  = fmaxf(mNew[r], sc[s][r]);
      }
    // Row max across the 16 lanes of each half-wave.
    float alpha[8];
#pragma unroll
    for (int r = 0; r < 8; ++r) {
#pragma unroll
      for (int off = 1; off < 16; off <<= 1)
        mNew[r] = fmaxf(mNew[r], __shfl_xor(mNew[r], off, 16));
      alpha[r] = __expf(mSt[r] - mNew[r]);
      mSt[r]   = mNew[r];
    }
    // P = exp(sc - m); row sums; stash P bf16 (C-layout -> A-layout via LDS).
    float rs[8];
#pragma unroll
    for (int r = 0; r < 8; ++r) rs[r] = 0.0f;
#pragma unroll
    for (int s = 0; s < 4; ++s)
#pragma unroll
      for (int r = 0; r < 8; ++r) {
        float p = __expf(sc[s][r] - mSt[r]);
        rs[r] += p;
        pT[wave][r + 8 * half][s * 16 + (lane & 15)] = (bf16)p;
      }
#pragma unroll
    for (int r = 0; r < 8; ++r) {
#pragma unroll
      for (int off = 1; off < 16; off <<= 1)
        rs[r] += __shfl_xor(rs[r], off, 16);
      lSt[r] = lSt[r] * alpha[r] + rs[r];
#pragma unroll
      for (int s = 0; s < 4; ++s) accO[s][r] *= alpha[r];
    }
    __syncthreads();   // P visible; all waves done reading kTs

    // O += P @ V : batch 2 A-frags + 8 B-frags, then 8 WMMAs.
    v16bf pA0 = load_a_frag(&pT[wave][0][0],  72, lane);
    v16bf pA1 = load_a_frag(&pT[wave][0][32], 72, lane);
    v16bf vb[4][2];
#pragma unroll
    for (int s = 0; s < 4; ++s) {
      vb[s][0] = load_b_fragT(&vTt[s * 16][0],  72, lane);
      vb[s][1] = load_b_fragT(&vTt[s * 16][32], 72, lane);
    }
#pragma unroll
    for (int s = 0; s < 4; ++s) accO[s] = wmma_bf16(pA0, vb[s][0], accO[s]);
#pragma unroll
    for (int s = 0; s < 4; ++s) accO[s] = wmma_bf16(pA1, vb[s][1], accO[s]);
    __syncthreads();   // done with tiles before next iteration overwrites
  }

  // out[b, q, h*64 + d] = O / l
#pragma unroll
  for (int s = 0; s < 4; ++s)
#pragma unroll
    for (int r = 0; r < 8; ++r) {
      const int row = qBase + r + 8 * half;
      const int col = cOff + s * 16 + (lane & 15);
      out[(rowOff + row) * D_MODEL + col] = accO[s][r] / lSt[r];
    }
}

extern "C" void kernel_launch(void* const* d_in, const int* in_sizes, int n_in,
                              void* d_out, int out_size, void* d_ws, size_t ws_size,
                              hipStream_t stream) {
  const float* hidden = (const float*)d_in[0];
  const int*   amask  = (const int*)d_in[1];
  const float* Wq     = (const float*)d_in[2];
  const float* bq     = (const float*)d_in[3];
  const float* Wk     = (const float*)d_in[4];
  const float* bk     = (const float*)d_in[5];
  float* out = (float*)d_out;

  bf16* qbf = (bf16*)d_ws;                                   // 8 MB
  bf16* kbf = qbf + (size_t)BATCH * SEQ * D_MODEL;           // 8 MB

  dim3 gProj(D_MODEL / 64, (BATCH * SEQ) / 64);              // 16 x 64
  qk_proj_kernel<<<gProj, 128, 0, stream>>>(hidden, Wq, bq, Wk, bk, qbf, kbf);

  dim3 gAttn(SEQ / 64, NHEADS, BATCH);                       // 32 x 16 x 2
  attn_kernel<<<gAttn, 128, 0, stream>>>(qbf, kbf, amask, out);
}
